// Head_37950331027555
// MI455X (gfx1250) — matrix operand order repaired and checked by reading, not compiled, not run on hardware
//
#include <hip/hip_runtime.h>

typedef __attribute__((ext_vector_type(16))) __bf16 bf16x16;
typedef __attribute__((ext_vector_type(8)))  __bf16 bf16x8;
typedef __attribute__((ext_vector_type(8)))  float  f32x8;

#define BB 8
#define TT 2048
#define CC 1024
#define DD 64
#define BT (BB * TT)

// Workspace layout (bf16 element offsets)
#define OFF_Q   ((size_t)0)
#define OFF_K   ((size_t)BT * DD)
#define OFF_VT  ((size_t)2 * BT * DD)                 // V transposed: [DD][BT]
#define OFF_WT  ((size_t)3 * BT * DD)                 // 3x transposed W: [DD][CC]
#define OFF_XBF (OFF_WT + (size_t)3 * DD * CC)        // x in bf16: [BT][CC]

union AFrag { bf16x16 v; bf16x8 h[2]; };

static __device__ __forceinline__ f32x8 wmma_bf16(bf16x16 a, bf16x16 b, f32x8 c) {
  // 8 args: (neg_a, A, neg_b, B, c_mod, C, reuse_a, reuse_b)
  return __builtin_amdgcn_wmma_f32_16x16x32_bf16(false, a, false, b, (short)0, c, false, false);
}

// ---------------------------------------------------------------------------
// Kernel 0a: x fp32 -> bf16 (vectorized, one-shot; removes cvt from GEMM loop)
// ---------------------------------------------------------------------------
__global__ void __launch_bounds__(256)
convert_x_kernel(const float* __restrict__ x, __bf16* __restrict__ xbf) {
  const size_t i = ((size_t)blockIdx.x * blockDim.x + threadIdx.x) * 8;
  const float4 f0 = *(const float4*)(x + i);
  const float4 f1 = *(const float4*)(x + i + 4);
  bf16x8 o;
  o[0] = (__bf16)f0.x; o[1] = (__bf16)f0.y; o[2] = (__bf16)f0.z; o[3] = (__bf16)f0.w;
  o[4] = (__bf16)f1.x; o[5] = (__bf16)f1.y; o[6] = (__bf16)f1.z; o[7] = (__bf16)f1.w;
  *(bf16x8*)(xbf + i) = o;
}

// ---------------------------------------------------------------------------
// Kernel 0b: W [CC,DD] fp32 -> W^T [DD,CC] bf16 (q pre-scaled by 1/sqrt(64))
// ---------------------------------------------------------------------------
__global__ void __launch_bounds__(256)
convert_w_kernel(const float* __restrict__ Wk, const float* __restrict__ Wq,
                 const float* __restrict__ Wv, __bf16* __restrict__ wt) {
  const int which = blockIdx.y;                       // 0=q, 1=k, 2=v
  const float* W  = (which == 0) ? Wq : (which == 1) ? Wk : Wv;
  const float sc  = (which == 0) ? 0.125f : 1.0f;
  const int t  = blockIdx.x * 256 + threadIdx.x;      // t = kk*DD + n
  const int kk = t / DD, n = t % DD;
  wt[(size_t)which * DD * CC + (size_t)n * CC + kk] = (__bf16)(W[t] * sc);
}

// ---------------------------------------------------------------------------
// Kernel 1: q/k/v = x @ W via WMMA.  One wave -> one 16-row tile, all 64
// columns, ALL THREE matrices (x read once). All fragment loads contiguous.
// q,k stored row-major [BT,DD]; v stored transposed [DD,BT].
// ---------------------------------------------------------------------------
__global__ void __launch_bounds__(256)
proj_qkv_kernel(const __bf16* __restrict__ xbf, const __bf16* __restrict__ wt,
                __bf16* __restrict__ ws)
{
  const int lane  = threadIdx.x & 31;
  const int wave  = threadIdx.x >> 5;
  const int mbase = (blockIdx.x * 8 + wave) * 16;
  const int lo = lane & 15, hi = lane >> 4;

  f32x8 acc[3][4];
  #pragma unroll
  for (int w = 0; w < 3; ++w)
    #pragma unroll
    for (int t = 0; t < 4; ++t)
      #pragma unroll
      for (int r = 0; r < 8; ++r) acc[w][t][r] = 0.0f;

  const __bf16* xrow = xbf + (size_t)(mbase + lo) * CC;

  for (int kb = 0; kb < CC; kb += 32) {
    // A fragment 16x32: lane row = mbase+lo; K = kb + 16*hi + [0,16) contiguous
    AFrag a;
    a.h[0] = *(const bf16x8*)(xrow + kb + hi * 16);
    a.h[1] = *(const bf16x8*)(xrow + kb + hi * 16 + 8);
    #pragma unroll
    for (int w = 0; w < 3; ++w) {
      const __bf16* wbase = wt + (size_t)w * DD * CC;
      #pragma unroll
      for (int t = 0; t < 4; ++t) {
        // B fragment 32x16: lane col n, K = kb + 16*hi + [0,16) contiguous in W^T
        const int n = t * 16 + lo;
        const bf16x16 bm = *(const bf16x16*)(wbase + (size_t)n * CC + kb + hi * 16);
        acc[w][t] = wmma_bf16(a.v, bm, acc[w][t]);
      }
    }
  }

  // C/D layout: VGPR r -> rows r (lanes 0-15) / r+8 (lanes 16-31), N = lo
  #pragma unroll
  for (int w = 0; w < 3; ++w)
    #pragma unroll
    for (int t = 0; t < 4; ++t)
      #pragma unroll
      for (int r = 0; r < 8; ++r) {
        const int row = mbase + r + 8 * hi;
        const int col = t * 16 + lo;
        if (w == 0)      ws[OFF_Q  + (size_t)row * DD + col] = (__bf16)acc[w][t][r];
        else if (w == 1) ws[OFF_K  + (size_t)row * DD + col] = (__bf16)acc[w][t][r];
        else             ws[OFF_VT + (size_t)col * BT + row] = (__bf16)acc[w][t][r];
      }
}

// ---------------------------------------------------------------------------
// Kernel 2: flash attention, one wave per (batch, 16-query tile), 32 keys per
// step: 4 WMMAs for S (two 16x16 sub-tiles), 4 WMMAs for O += P@V (full K=32).
// ---------------------------------------------------------------------------
__global__ void __launch_bounds__(32)
flash_attn_kernel(const __bf16* __restrict__ ws, float* __restrict__ out)
{
  const int lane  = threadIdx.x & 31;
  const int mt    = blockIdx.x;
  const int b     = blockIdx.y;
  const int qbase = mt * 16;
  const int nkeys = qbase + 16;
  const int lo = lane & 15, hi = lane >> 4;

  const __bf16* q  = ws + OFF_Q;
  const __bf16* k  = ws + OFF_K;
  const __bf16* vt = ws + OFF_VT;
  const size_t rowoff = (size_t)b * TT;

  // Q A-fragments for K = [0,32) and [32,64) -- contiguous 16B chunks
  AFrag qa[2];
  {
    const __bf16* qrow = q + (rowoff + qbase + lo) * DD;
    #pragma unroll
    for (int h = 0; h < 2; ++h) {
      qa[h].h[0] = *(const bf16x8*)(qrow + h * 32 + hi * 16);
      qa[h].h[1] = *(const bf16x8*)(qrow + h * 32 + hi * 16 + 8);
    }
  }

  float rowmax[8], rowsum[8];
  f32x8 o[4];
  #pragma unroll
  for (int r = 0; r < 8; ++r) { rowmax[r] = -__builtin_inff(); rowsum[r] = 0.0f; }
  #pragma unroll
  for (int t = 0; t < 4; ++t)
    #pragma unroll
    for (int r = 0; r < 8; ++r) o[t][r] = 0.0f;

  __shared__ __bf16 lds_p[16 * 32];   // one wave per block: P staging tile

  for (int s0 = 0; s0 < nkeys; s0 += 32) {
    const bool have2 = (s0 + 16 < nkeys);    // wave-uniform: 2nd sub-tile valid

    // prefetch next key block (global_prefetch_b8); stays inside workspace
    __builtin_prefetch(k + (rowoff + s0 + 32 + lo) * DD, 0, 0);

    // ---- S sub-tile 1: keys [s0, s0+16) ----
    f32x8 s1, s2;
    #pragma unroll
    for (int r = 0; r < 8; ++r) s1[r] = 0.0f;
    #pragma unroll
    for (int h = 0; h < 2; ++h) {
      const bf16x16 kf = *(const bf16x16*)(k + (rowoff + s0 + lo) * DD + h * 32 + hi * 16);
      s1 = wmma_bf16(qa[h].v, kf, s1);
    }
    // ---- S sub-tile 2: keys [s0+16, s0+32) (uniform branch, EXEC stays full)
    if (have2) {
      #pragma unroll
      for (int r = 0; r < 8; ++r) s2[r] = 0.0f;
      #pragma unroll
      for (int h = 0; h < 2; ++h) {
        const bf16x16 kf = *(const bf16x16*)(k + (rowoff + s0 + 16 + lo) * DD + h * 32 + hi * 16);
        s2 = wmma_bf16(qa[h].v, kf, s2);
      }
    } else {
      #pragma unroll
      for (int r = 0; r < 8; ++r) s2[r] = -__builtin_inff();
    }

    // ---- causal mask (only steps overlapping the diagonal) ----
    if (s0 + 31 > qbase) {
      #pragma unroll
      for (int r = 0; r < 8; ++r) {
        const int mg = qbase + r + 8 * hi;
        if (s0 + lo > mg)      s1[r] = -__builtin_inff();
        if (s0 + 16 + lo > mg) s2[r] = -__builtin_inff();
      }
    }

    // ---- online softmax over 32 columns (16-lane half-group reductions) ----
    float p1[8], p2[8];
    #pragma unroll
    for (int r = 0; r < 8; ++r) {
      float tmax = fmaxf(s1[r], s2[r]);
      #pragma unroll
      for (int off = 8; off >= 1; off >>= 1)
        tmax = fmaxf(tmax, __shfl_xor(tmax, off, 32));
      const float mnew = fmaxf(rowmax[r], tmax);
      const float corr = __expf(rowmax[r] - mnew);
      p1[r] = __expf(s1[r] - mnew);
      p2[r] = __expf(s2[r] - mnew);
      float psum = p1[r] + p2[r];
      #pragma unroll
      for (int off = 8; off >= 1; off >>= 1)
        psum += __shfl_xor(psum, off, 32);
      rowsum[r] = rowsum[r] * corr + psum;
      rowmax[r] = mnew;
      #pragma unroll
      for (int t4 = 0; t4 < 4; ++t4) o[t4][r] *= corr;
    }

    // ---- P: C-layout -> A-layout via LDS (16x32 bf16) ----
    #pragma unroll
    for (int r = 0; r < 8; ++r) {
      const int row = r + 8 * hi;
      lds_p[row * 32 + lo]      = (__bf16)p1[r];
      lds_p[row * 32 + 16 + lo] = (__bf16)p2[r];
    }
    __asm__ volatile("s_wait_dscnt 0" ::: "memory");
    AFrag pa;
    pa.h[0] = *(const bf16x8*)(lds_p + lo * 32 + hi * 16);
    pa.h[1] = *(const bf16x8*)(lds_p + lo * 32 + hi * 16 + 8);

    // ---- O += P @ V : B fragment K rows = s0 + 16*hi + [0,16), col n ----
    #pragma unroll
    for (int t4 = 0; t4 < 4; ++t4) {
      bf16x16 vf;
      if (hi == 0 || have2) {
        vf = *(const bf16x16*)(vt + (size_t)(t4 * 16 + lo) * BT + rowoff + s0 + 16 * hi);
      } else {
        #pragma unroll
        for (int i = 0; i < 16; ++i) vf[i] = (__bf16)0.0f;
      }
      o[t4] = wmma_bf16(pa.v, vf, o[t4]);
    }
  }

  // ---- epilogue: normalize, write fp32 ----
  #pragma unroll
  for (int t4 = 0; t4 < 4; ++t4)
    #pragma unroll
    for (int r = 0; r < 8; ++r) {
      const int row = qbase + r + 8 * hi;
      const int col = t4 * 16 + lo;
      out[(rowoff + row) * DD + col] = o[t4][r] / rowsum[r];
    }
}

// ---------------------------------------------------------------------------
extern "C" void kernel_launch(void* const* d_in, const int* in_sizes, int n_in,
                              void* d_out, int out_size, void* d_ws, size_t ws_size,
                              hipStream_t stream) {
  const float* x  = (const float*)d_in[0];
  const float* Wk = (const float*)d_in[1];
  const float* Wq = (const float*)d_in[2];
  const float* Wv = (const float*)d_in[3];
  __bf16* ws = (__bf16*)d_ws;   // ~38.5 MB used
  float* out = (float*)d_out;

  convert_x_kernel<<<(size_t)BT * CC / 8 / 256, 256, 0, stream>>>(x, ws + OFF_XBF);
  dim3 wgrid(CC * DD / 256, 3);
  convert_w_kernel<<<wgrid, 256, 0, stream>>>(Wk, Wq, Wv, ws + OFF_WT);

  proj_qkv_kernel<<<BT / 16 / 8, 256, 0, stream>>>(ws + OFF_XBF, ws + OFF_WT, ws);

  dim3 agrid(TT / 16, BB);
  flash_attn_kernel<<<agrid, 32, 0, stream>>>(ws, out);
}
